// Head_29575144800617
// MI455X (gfx1250) — matrix operand order repaired and checked by reading, not compiled, not run on hardware
//
#include <hip/hip_runtime.h>
#include <hip/hip_bf16.h>
#include <stdint.h>

typedef __attribute__((ext_vector_type(16))) __bf16 v16bf;
typedef __attribute__((ext_vector_type(8)))  float  v8f;
typedef __attribute__((ext_vector_type(8)))  unsigned int v8u;
typedef __attribute__((ext_vector_type(4)))  unsigned int u32x4;
typedef __attribute__((ext_vector_type(8)))  int i32x8;
typedef __attribute__((ext_vector_type(4)))  int i32x4;

#define Bv 4
#define Tv 2048
#define Ev 1024
#define Hv 128
#define BT (Bv*Tv)   // 8192

#if defined(__HIP_DEVICE_COMPILE__) && defined(__has_builtin)
# if __has_builtin(__builtin_amdgcn_tensor_load_to_lds) && __has_builtin(__builtin_amdgcn_s_wait_tensorcnt)
#  define USE_TDM 1
# else
#  define USE_TDM 0
# endif
#else
# define USE_TDM 0
#endif

// ---------------------------------------------------------------------------
// Kernel 1: q/k/v projections (unchanged structure: emits packed bf16 cvts +
// v_wmma_f32_16x16x32_bf16). q,k bf16 [BT][H]; v bf16 transposed [B][H][T].
// ---------------------------------------------------------------------------
__global__ __launch_bounds__(256) void qkv_kernel(
    const float* __restrict__ x,
    const float* __restrict__ Wq, const float* __restrict__ bq,
    const float* __restrict__ Wk, const float* __restrict__ bk,
    const float* __restrict__ Wv, const float* __restrict__ bv,
    __bf16* __restrict__ qout, __bf16* __restrict__ kout, __bf16* __restrict__ vT)
{
    const int wid  = blockIdx.x * (blockDim.x >> 5) + (threadIdx.x >> 5);
    const int lane = threadIdx.x & 31;
    const int hi   = lane >> 4;
    const int ln   = lane & 15;

    const int mat = wid / (512 * 8);         // 0=q, 1=k, 2=v (blocks never straddle)
    const int rem = wid % (512 * 8);
    const int mt  = rem >> 3;
    const int nt  = rem & 7;

    const float* W    = (mat == 0) ? Wq : ((mat == 1) ? Wk : Wv);
    const float* bias = (mat == 0) ? bq : ((mat == 1) ? bk : bv);

    const int arow = mt * 16 + ln;           // A row (M = lane%16)
    const int ncol = nt * 16 + ln;           // B/C column (N = lane%16)

    v8f acc = {};
    const float* xr = x + (size_t)arow * Ev;
    for (int k0 = 0; k0 < Ev; k0 += 32) {
        v16bf a, b;
        #pragma unroll
        for (int j = 0; j < 4; ++j) {        // A: VGPR j -> K=k0+8hi+2j ; VGPR 4+j -> +16
            float2 p0 = *(const float2*)(xr + k0 +      8 * hi + 2 * j);
            float2 p1 = *(const float2*)(xr + k0 + 16 + 8 * hi + 2 * j);
            a[2*j]     = (__bf16)p0.x;  a[2*j + 1]     = (__bf16)p0.y;
            a[8 + 2*j] = (__bf16)p1.x;  a[8 + 2*j + 1] = (__bf16)p1.y;
        }
        #pragma unroll
        for (int j = 0; j < 8; ++j) {        // B: VGPR j -> K=k0+8hi+2j ; VGPR 4+j -> +16
            const int kk = k0 + 8 * hi + 2 * j + 8 * (j >> 2);
            b[2*j]     = (__bf16)W[(size_t)kk * Hv + ncol];
            b[2*j + 1] = (__bf16)W[(size_t)(kk + 1) * Hv + ncol];
        }
        acc = __builtin_amdgcn_wmma_f32_16x16x32_bf16(
                  false, a, false, b, (short)0, acc, false, false);
    }

    const float bb    = bias[ncol];
    const float scale = (mat == 0) ? 0.08838834764831845f : 1.0f; // H^-0.5 into q
    #pragma unroll
    for (int r = 0; r < 8; ++r) {
        const float val  = (acc[r] + bb) * scale;
        const int   grow = mt * 16 + 8 * hi + r;
        if (mat == 2) {
            const int bb_ = grow >> 11, t = grow & (Tv - 1);
            vT[((size_t)bb_ * Hv + ncol) * Tv + t] = (__bf16)val;
        } else {
            __bf16* dst = (mat == 0) ? qout : kout;
            dst[(size_t)grow * Hv + ncol] = (__bf16)val;
        }
    }
}

// ---------------------------------------------------------------------------
// TDM: 2D tile (tile_w x tile_h elements of 2 bytes) -> LDS, per D# spec.
// group0: count=1 | lds_addr | global_addr(57b) | type=2
// group1: data_size=1(2B) | tensor dims | tile dims | dim0 stride
// ---------------------------------------------------------------------------
#if USE_TDM
__device__ inline void tdm_load_2d(uint32_t lds_off, const void* gp,
                                   uint32_t tile_w, uint32_t tile_h,
                                   uint32_t tensor_w, uint32_t tensor_h,
                                   uint32_t row_stride_elems)
{
    const uint64_t ga = (uint64_t)(uintptr_t)gp;
    u32x4 g0;
    g0[0] = 1u;                                                   // count=1
    g0[1] = lds_off;                                              // LDS byte addr
    g0[2] = (uint32_t)ga;                                         // global addr lo
    g0[3] = (uint32_t)((ga >> 32) & 0x01FFFFFFu) | (2u << 30);    // addr hi | type=2
    i32x8 g1;
    g1[0] = (int)(1u << 16);                                      // mask=0, data_size=2B
    g1[1] = (int)((tensor_w & 0xFFFFu) << 16);                    // tensor_dim0 lo16
    g1[2] = (int)(((tensor_w >> 16) & 0xFFFFu) | ((tensor_h & 0xFFFFu) << 16));
    g1[3] = (int)(((tensor_h >> 16) & 0xFFFFu) | ((tile_w & 0xFFFFu) << 16));
    g1[4] = (int)(tile_h & 0xFFFFu);                              // tile_dim1, tile_dim2=0
    g1[5] = (int)row_stride_elems;                                // tensor_dim0_stride lo32
    g1[6] = 0;
    g1[7] = 0;
    const i32x4 z4 = {0, 0, 0, 0};
#if defined(__clang_major__) && (__clang_major__ >= 23)
    const i32x8 z8 = {0, 0, 0, 0, 0, 0, 0, 0};
    __builtin_amdgcn_tensor_load_to_lds(g0, g1, z4, z4, z8, 0);
#else
    __builtin_amdgcn_tensor_load_to_lds(g0, g1, z4, z4, 0);
#endif
}
#endif

// ---------------------------------------------------------------------------
// Kernel 2: causal flash attention. Block = 8 waves = 8 adjacent query tiles
// of one batch. K/V key-blocks of 32 staged once per block into LDS (TDM,
// double-buffered, wave0 issues + s_wait_tensorcnt). Fragments re-read as
// contiguous 16B LDS loads. 16 WMMAs per key block per wave.
// ---------------------------------------------------------------------------
__global__ __launch_bounds__(256) void attn_kernel(
    const __bf16* __restrict__ qg, const __bf16* __restrict__ kg,
    const __bf16* __restrict__ vT, float* __restrict__ out)
{
    __shared__ __bf16 kbuf[2][32 * Hv];     // [t=32][h=128]  2 x 8KB
    __shared__ __bf16 vbuf[2][Hv * 32];     // [h=128][t=32]  2 x 8KB
    __shared__ __bf16 pstage[8][16 * 32];   // per-wave P relayout, 8 x 1KB

    const int tid   = threadIdx.x;
    const int wslot = tid >> 5;
    const int lane  = tid & 31;
    const int hi    = lane >> 4;
    const int ln    = lane & 15;

    const int batch = blockIdx.x >> 4;            // 16 blocks per batch
    const int qt0   = (blockIdx.x & 15) * 8;
    const int qbase = (qt0 + wslot) * 16;

    // Preload Q: 4 A-fragments (K=32 chunks over H=128), 16B global loads.
    v16bf qa[4];
    {
        const uint4* q4  = (const uint4*)qg;
        const size_t row = (size_t)batch * Tv + qbase + ln;
        #pragma unroll
        for (int c = 0; c < 4; ++c) {
            uint4 a0 = q4[row * 16 + c * 4 + hi];        // kk = 8hi + [0..7]
            uint4 a1 = q4[row * 16 + c * 4 + 2 + hi];    // kk = 16 + 8hi + [0..7]
            v8u u = {a0.x, a0.y, a0.z, a0.w, a1.x, a1.y, a1.z, a1.w};
            qa[c] = __builtin_bit_cast(v16bf, u);
        }
    }

    v8f zero = {};
    v8f oacc[8];
    #pragma unroll
    for (int i = 0; i < 8; ++i) oacc[i] = zero;
    float mrow[8], lrow[8];
    #pragma unroll
    for (int r = 0; r < 8; ++r) { mrow[r] = -1e30f; lrow[r] = 0.f; }

    const int mykb  = (qbase + 15) >> 5;            // this wave's last key block
    const int kbmax = (qt0 * 16 + 127) >> 5;        // block-wide last key block

    // ---- per-wave compute on staged tiles (EXEC all-ones throughout) ----
    auto compute = [&](int kb, int cur) {
        const __bf16* kt = &kbuf[cur][0];
        const __bf16* vt = &vbuf[cur][0];
        v8f sf[2];
        #pragma unroll
        for (int s = 0; s < 2; ++s) {
            v8f sacc = zero;
            #pragma unroll
            for (int c = 0; c < 4; ++c) {
                const uint4* b4 = (const uint4*)(kt + ((s * 16 + ln) * Hv + c * 32 + 8 * hi));
                uint4 lo = b4[0], h4 = b4[2];   // kk = 8hi + [0..7], 16 + 8hi + [0..7]
                v8u u = {lo.x, lo.y, lo.z, lo.w, h4.x, h4.y, h4.z, h4.w};
                v16bf bf = __builtin_bit_cast(v16bf, u);
                sacc = __builtin_amdgcn_wmma_f32_16x16x32_bf16(
                           false, qa[c], false, bf, (short)0, sacc, false, false);
            }
            sf[s] = sacc;
        }
        // causal mask + online softmax (row stats within 16-lane halves)
        float al[8];
        #pragma unroll
        for (int r = 0; r < 8; ++r) {
            const int qi  = qbase + 8 * hi + r;
            const int kj0 = kb * 32 + ln;
            float s0 = (kj0      <= qi) ? sf[0][r] : -1e30f;
            float s1 = (kj0 + 16 <= qi) ? sf[1][r] : -1e30f;
            float t = fmaxf(s0, s1);
            #pragma unroll
            for (int m = 1; m < 16; m <<= 1) t = fmaxf(t, __shfl_xor(t, m, 32));
            const float mnew  = fmaxf(mrow[r], t);
            const float alpha = __expf(mrow[r] - mnew);
            mrow[r] = mnew;  al[r] = alpha;
            const float p0 = __expf(s0 - mnew);
            const float p1 = __expf(s1 - mnew);
            sf[0][r] = p0;  sf[1][r] = p1;
            float ps = p0 + p1;
            #pragma unroll
            for (int m = 1; m < 16; m <<= 1) ps += __shfl_xor(ps, m, 32);
            lrow[r] = lrow[r] * alpha + ps;
        }
        #pragma unroll
        for (int n = 0; n < 8; ++n)
            #pragma unroll
            for (int r = 0; r < 8; ++r) oacc[n][r] *= al[r];

        // relayout P (C-layout -> A-fragment) via per-wave LDS bounce
        __bf16* pb = &pstage[wslot][0];
        #pragma unroll
        for (int r = 0; r < 8; ++r) {
            const int m = 8 * hi + r;
            pb[m * 32 +      ln] = (__bf16)sf[0][r];
            pb[m * 32 + 16 + ln] = (__bf16)sf[1][r];
        }
        asm volatile("s_wait_dscnt 0" ::: "memory");
        v16bf pa;
        {
            const uint4* p4 = (const uint4*)pb;
            uint4 a0 = p4[ln * 4 + hi];
            uint4 a1 = p4[ln * 4 + 2 + hi];
            v8u u = {a0.x, a0.y, a0.z, a0.w, a1.x, a1.y, a1.z, a1.w};
            pa = __builtin_bit_cast(v16bf, u);
        }
        asm volatile("" ::: "memory");

        // O += P @ V : 8 h-tiles, contiguous 16B LDS loads from vbuf
        #pragma unroll
        for (int n = 0; n < 8; ++n) {
            const uint4* b4 = (const uint4*)(vt + ((n * 16 + ln) * 32 + 8 * hi));
            uint4 lo = b4[0], h4 = b4[2];
            v8u u = {lo.x, lo.y, lo.z, lo.w, h4.x, h4.y, h4.z, h4.w};
            v16bf vf = __builtin_bit_cast(v16bf, u);
            oacc[n] = __builtin_amdgcn_wmma_f32_16x16x32_bf16(
                          false, pa, false, vf, (short)0, oacc[n], false, false);
        }
    };

#if USE_TDM
    auto stage = [&](int kb, int cur) {      // issued by wave 0 only (scalar branch)
        tdm_load_2d((uint32_t)(uintptr_t)&kbuf[cur][0],
                    kg + ((size_t)batch * Tv + (size_t)kb * 32) * Hv,
                    /*tile_w*/Hv, /*tile_h*/32, /*tensor_w*/Hv, /*tensor_h*/32,
                    /*stride*/Hv);
        tdm_load_2d((uint32_t)(uintptr_t)&vbuf[cur][0],
                    vT + (size_t)batch * Hv * Tv + (size_t)kb * 32,
                    /*tile_w*/32, /*tile_h*/Hv, /*tensor_w*/32, /*tensor_h*/Hv,
                    /*stride*/Tv);
    };
    if (__builtin_amdgcn_readfirstlane(wslot) == 0) {
        stage(0, 0);
        __builtin_amdgcn_s_wait_tensorcnt(0);
    }
    __syncthreads();
    for (int kb = 0; kb <= kbmax; ++kb) {
        const int cur = kb & 1;
        if (kb < kbmax && __builtin_amdgcn_readfirstlane(wslot) == 0)
            stage(kb + 1, cur ^ 1);                       // prefetch next tile
        if (__builtin_amdgcn_readfirstlane((int)(kb <= mykb)))
            compute(kb, cur);
        if (kb < kbmax && __builtin_amdgcn_readfirstlane(wslot) == 0)
            __builtin_amdgcn_s_wait_tensorcnt(0);         // next buffer ready
        __syncthreads();
    }
#else
    for (int kb = 0; kb <= kbmax; ++kb) {
        {   // cooperative staging fallback (256 threads)
            const uint4* ks = (const uint4*)(kg + ((size_t)batch * Tv + (size_t)kb * 32) * Hv);
            uint4* kd = (uint4*)&kbuf[0][0];
            for (int i = tid; i < (32 * Hv) / 8; i += 256) kd[i] = ks[i];
            uint4* vd = (uint4*)&vbuf[0][0];
            for (int i = tid; i < (Hv * 32) / 8; i += 256) {
                const int row = i >> 2, q = i & 3;
                vd[i] = ((const uint4*)(vT + ((size_t)batch * Hv + row) * Tv + (size_t)kb * 32))[q];
            }
        }
        __syncthreads();
        if (__builtin_amdgcn_readfirstlane((int)(kb <= mykb)))
            compute(kb, 0);
        __syncthreads();
    }
#endif

    // ---- epilogue ----
    #pragma unroll
    for (int n = 0; n < 8; ++n) {
        #pragma unroll
        for (int r = 0; r < 8; ++r) {
            const int m = 8 * hi + r;
            out[((size_t)batch * Tv + qbase + m) * Hv + n * 16 + ln] =
                oacc[n][r] / lrow[r];
        }
    }
}

// ---------------------------------------------------------------------------
extern "C" void kernel_launch(void* const* d_in, const int* in_sizes, int n_in,
                              void* d_out, int out_size, void* d_ws, size_t ws_size,
                              hipStream_t stream) {
    const float* x  = (const float*)d_in[0];
    const float* Wq = (const float*)d_in[1];
    const float* bq = (const float*)d_in[2];
    const float* Wk = (const float*)d_in[3];
    const float* bk = (const float*)d_in[4];
    const float* Wv = (const float*)d_in[5];
    const float* bv = (const float*)d_in[6];

    __bf16* qb = (__bf16*)d_ws;                    // 2 MB
    __bf16* kb = qb + (size_t)BT * Hv;             // 2 MB
    __bf16* vT = kb + (size_t)BT * Hv;             // 2 MB, [B][H][T]

    qkv_kernel<<<1536, 256, 0, stream>>>(x, Wq, bq, Wk, bk, Wv, bv, qb, kb, vT);
    attn_kernel<<<64, 256, 0, stream>>>(qb, kb, vT, (float*)d_out);
}
